// LigColumnAttention_36172214566950
// MI455X (gfx1250) — compile-verified, matrix operand order checked by reading, not run on hardware
//
#include <hip/hip_runtime.h>
#include <hip/hip_bf16.h>

typedef __attribute__((ext_vector_type(16))) _Float16 v16h;
typedef __attribute__((ext_vector_type(8)))  _Float16 v8h;
typedef __attribute__((ext_vector_type(8)))  float    v8f;

#define S_LEN 256
#define R_LEN 384
#define C_DIM 256
#define H_NUM 8
#define D_DIM 32

#define NTHREADS 512   // 16 wave32 -> 4 waves per SIMD32 (latency hiding)
#define NWAVES   16
#define ROWS_PW  16    // query/sequence rows owned by each wave

// LDS layout (offsets in halves; total 163840 halves = 327680 B = 320 KB)
#define LDS_X   0        // x  [256][256] f16  (layernormed activations)
#define LDS_O   65536    // o  [256][256] f16  (gated per-head outputs)
#define LDS_WT  131072   // wt [32][256]  f16  (transposed weight tile; P-staging overlay)
#define LDS_Q   139264   // q  [256][32]  f16
#define LDS_K   147456   // k  [256][32]  f16
#define LDS_VT  155648   // vT [32][256]  f16
#define LDS_TOTAL_BYTES 327680

// Load a 16x32 f16 WMMA fragment (A-layout, or B-layout when the matrix
// columns are stored contiguously as rows). Per ISA 7.12.2: lanes 0-15 hold
// row (lane&15), K=0..7 (VGPR0-3) and K=16..23 (VGPR4-7); lanes 16-31 hold
// K=8..15 and K=24..31. Two contiguous 16B chunks per lane -> ds_load_b128.
__device__ inline v16h ldfrag(const _Float16* p, int rstride) {
  const int lane = threadIdx.x & 31;
  const _Float16* q = p + (lane & 15) * rstride + ((lane >> 4) << 3);
  v8h lo = *(const v8h*)q;
  v8h hi = *(const v8h*)(q + 16);
  v16h r;
#pragma unroll
  for (int i = 0; i < 8; ++i) { r[i] = lo[i]; r[i + 8] = hi[i]; }
  return r;
}

__device__ inline v8f wmma16(v16h a, v16h b, v8f c) {
  return __builtin_amdgcn_wmma_f32_16x16x32_f16(false, a, false, b, (short)0, c,
                                                false, false);
}

// Stage W[:, h*32 : h*32+32] transposed into LDS: wt[d][c], c contiguous.
__device__ inline void load_wT(_Float16* wt, const float* __restrict__ w, int h) {
  const int tid = threadIdx.x;
  const int d = tid & 31;
  const int c0 = tid >> 5;            // 0..15
#pragma unroll 4
  for (int c = c0; c < C_DIM; c += NWAVES)
    wt[d * 256 + c] = (_Float16)w[c * 256 + h * 32 + d];
}

// Per-wave projection: this wave's 16 rows of (x @ wtT_block) -> [16 x 32].
// transposed=false -> dst[s][32] row-major (stride 32); true -> dst[d][256].
__device__ inline void project(const _Float16* xs, const _Float16* wt,
                               _Float16* dst, int rowb, int hl, int ln16,
                               float scale, bool transposed) {
#pragma unroll
  for (int nt = 0; nt < 2; ++nt) {
    v8f acc = {};
#pragma unroll
    for (int kc = 0; kc < C_DIM; kc += 32) {
      v16h a = ldfrag(xs + rowb * 256 + kc, 256);
      v16h b = ldfrag(wt + nt * 16 * 256 + kc, 256);
      acc = wmma16(a, b, acc);
    }
#pragma unroll
    for (int i = 0; i < 8; ++i) {
      const int srow = rowb + i + 8 * hl;   // C-layout: M = i + 8*(lane>>4)
      const int dcol = nt * 16 + ln16;      //           N = lane & 15
      const _Float16 val = (_Float16)(acc[i] * scale);
      if (transposed) dst[dcol * 256 + srow] = val;
      else            dst[srow * 32 + dcol] = val;
    }
  }
}

__global__ void __launch_bounds__(NTHREADS, 1)
col_attn_fused(const float* __restrict__ msa_act,
               const float* __restrict__ msa_mask,
               const float* __restrict__ ln_w, const float* __restrict__ ln_b,
               const float* __restrict__ wq, const float* __restrict__ wk,
               const float* __restrict__ wvp, const float* __restrict__ wg,
               const float* __restrict__ bg, const float* __restrict__ wo,
               const float* __restrict__ bo, float* __restrict__ out) {
  extern __shared__ _Float16 smem[];
  _Float16* xs = smem + LDS_X;
  _Float16* ob = smem + LDS_O;
  _Float16* wt = smem + LDS_WT;
  _Float16* qh = smem + LDS_Q;
  _Float16* kh = smem + LDS_K;
  _Float16* vt = smem + LDS_VT;

  const int r     = blockIdx.x;        // residue: the "batch" dim of col-attn
  const int tid   = threadIdx.x;
  const int lane  = tid & 31;
  const int wv_id = tid >> 5;          // 16 waves; wave w owns s rows [16w,16w+16)
  const int hl    = lane >> 4;
  const int ln16  = lane & 15;
  const int rowb  = wv_id * ROWS_PW;

  // ---------------- Phase 1: LayerNorm into LDS (f16) --------------------
  float lnw[8], lnbv[8];
#pragma unroll
  for (int i = 0; i < 8; ++i) { lnw[i] = ln_w[lane * 8 + i]; lnbv[i] = ln_b[lane * 8 + i]; }

  for (int sr = 0; sr < ROWS_PW; ++sr) {
    const int s = rowb + sr;
    const float* row = msa_act + ((size_t)s * R_LEN + r) * C_DIM;
    float v[8];
#pragma unroll
    for (int i = 0; i < 8; ++i) v[i] = row[lane * 8 + i];
    float sum = 0.f, sq = 0.f;
#pragma unroll
    for (int i = 0; i < 8; ++i) { sum += v[i]; sq += v[i] * v[i]; }
#pragma unroll
    for (int m = 16; m >= 1; m >>= 1) {
      sum += __shfl_xor(sum, m, 32);
      sq  += __shfl_xor(sq, m, 32);
    }
    const float mu   = sum * (1.f / 256.f);
    const float var  = sq * (1.f / 256.f) - mu * mu;
    const float rstd = rsqrtf(var + 1e-5f);
    v8h hx;
#pragma unroll
    for (int i = 0; i < 8; ++i)
      hx[i] = (_Float16)((v[i] - mu) * rstd * lnw[i] + lnbv[i]);
    *(v8h*)(xs + s * 256 + lane * 8) = hx;
  }

  // Per-lane attention bias for the 16 key-tiles this lane sees in C-layout.
  float kbias[16];
#pragma unroll
  for (int kt = 0; kt < 16; ++kt)
    kbias[kt] = 1e9f * (msa_mask[(size_t)(kt * 16 + ln16) * R_LEN + r] - 1.f);

  __syncthreads();

  v8f oaccv[2];  // per-wave O accumulators [ntile]; persist through gating

  // ---------------- Phase 2: per-head fused attention ---------------------
  for (int h = 0; h < H_NUM; ++h) {
    __builtin_prefetch(wk + h * 32, 0, 0);   // global_prefetch_b8
    load_wT(wt, wq, h);
    __syncthreads();
    project(xs, wt, qh, rowb, hl, ln16, 0.17677669529663688f /*1/sqrt(32)*/, false);
    __syncthreads();

    __builtin_prefetch(wvp + h * 32, 0, 0);
    load_wT(wt, wk, h);
    __syncthreads();
    project(xs, wt, kh, rowb, hl, ln16, 1.f, false);
    __syncthreads();

    load_wT(wt, wvp, h);
    __syncthreads();
    project(xs, wt, vt, rowb, hl, ln16, 1.f, true);  // store V transposed
    __syncthreads();  // all q/k/vT visible; wt now free -> P staging overlay

    // ---- flash-style attention, 16 query rows per wave -------------------
    _Float16* stg = wt + wv_id * 512;  // per-wave [16 q][32 k] f16 staging
    float rmax[8], rsum[8];
#pragma unroll
    for (int i = 0; i < 8; ++i) { rmax[i] = -1e30f; rsum[i] = 0.f; }
    oaccv[0] = (v8f){};
    oaccv[1] = (v8f){};
    const v16h qa = ldfrag(qh + rowb * 32, 32);  // D=32 = exactly one WMMA K

    for (int kp = 0; kp < 8; ++kp) {  // 32 keys per step
      const int key0 = kp * 32;
      v8f s0 = {}, s1 = {};
      s0 = wmma16(qa, ldfrag(kh + (key0 + 0)  * 32, 32), s0);
      s1 = wmma16(qa, ldfrag(kh + (key0 + 16) * 32, 32), s1);
#pragma unroll
      for (int i = 0; i < 8; ++i) {
        const float a = s0[i] + kbias[2 * kp];
        const float b = s1[i] + kbias[2 * kp + 1];
        float t = fmaxf(a, b);
#pragma unroll
        for (int m = 1; m <= 8; m <<= 1) t = fmaxf(t, __shfl_xor(t, m, 32));
        const float nm = fmaxf(rmax[i], t);
        const float sc = __expf(rmax[i] - nm);
        rmax[i] = nm;
        const float pa = __expf(a - nm);
        const float pb = __expf(b - nm);
        float ps = pa + pb;
#pragma unroll
        for (int m = 1; m <= 8; m <<= 1) ps += __shfl_xor(ps, m, 32);
        rsum[i] = rsum[i] * sc + ps;
        oaccv[0][i] *= sc;
        oaccv[1][i] *= sc;
        const int prow = i + 8 * hl;             // C-layout scatter of P
        stg[prow * 32 + ln16]      = (_Float16)pa;
        stg[prow * 32 + 16 + ln16] = (_Float16)pb;
      }
      const v16h pf = ldfrag(stg, 32);            // P as A-fragment
      oaccv[0] = wmma16(pf, ldfrag(vt + 0 * 256 + key0, 256), oaccv[0]);
      oaccv[1] = wmma16(pf, ldfrag(vt + 16 * 256 + key0, 256), oaccv[1]);
    }
#pragma unroll
    for (int nt = 0; nt < 2; ++nt)
#pragma unroll
      for (int i = 0; i < 8; ++i)
        oaccv[nt][i] *= __builtin_amdgcn_rcpf(rsum[i]);

    // ---- gating: g = sigmoid(x @ wg_h + bg_h); store gated O -------------
    __syncthreads();           // staging done before wt is reloaded
    load_wT(wt, wg, h);
    __syncthreads();
#pragma unroll
    for (int nt = 0; nt < 2; ++nt) {
      v8f acc = {};
#pragma unroll
      for (int kc = 0; kc < C_DIM; kc += 32) {
        v16h a = ldfrag(xs + rowb * 256 + kc, 256);
        v16h b = ldfrag(wt + nt * 16 * 256 + kc, 256);
        acc = wmma16(a, b, acc);
      }
      const float bgv = bg[h * 32 + nt * 16 + ln16];
#pragma unroll
      for (int i = 0; i < 8; ++i) {
        const float g = __builtin_amdgcn_rcpf(1.f + __expf(-(acc[i] + bgv)));
        const int srow = rowb + i + 8 * hl;
        ob[srow * 256 + h * 32 + nt * 16 + ln16] =
            (_Float16)(oaccv[nt][i] * g);
      }
    }
    __syncthreads();           // ob block written; wt free for next head
  }

  // ---------------- Phase 3: out = o @ wo + bo ----------------------------
  for (int cb = 0; cb < 8; ++cb) {  // 32 output channels per chunk
    {  // stage wo[:, cb*32 : cb*32+32] transposed: wt[c_local][hd]
      const int d = tid & 31;
      const int k0 = tid >> 5;
#pragma unroll 4
      for (int k = k0; k < 256; k += NWAVES)
        wt[d * 256 + k] = (_Float16)wo[k * 256 + cb * 32 + d];
    }
    __syncthreads();
#pragma unroll
    for (int nt = 0; nt < 2; ++nt) {
      v8f acc = {};
#pragma unroll
      for (int kc = 0; kc < 256; kc += 32) {
        v16h a = ldfrag(ob + rowb * 256 + kc, 256);
        v16h b = ldfrag(wt + nt * 16 * 256 + kc, 256);
        acc = wmma16(a, b, acc);
      }
      const int c = cb * 32 + nt * 16 + ln16;
      const float bov = bo[c];
#pragma unroll
      for (int i = 0; i < 8; ++i) {
        const int s = rowb + i + 8 * hl;
        out[((size_t)s * R_LEN + r) * C_DIM + c] = acc[i] + bov;
      }
    }
    __syncthreads();   // wt reload for next channel chunk
  }
}

extern "C" void kernel_launch(void* const* d_in, const int* in_sizes, int n_in,
                              void* d_out, int out_size, void* d_ws, size_t ws_size,
                              hipStream_t stream) {
  (void)in_sizes; (void)n_in; (void)out_size; (void)d_ws; (void)ws_size;
  const float* msa_act  = (const float*)d_in[0];
  const float* msa_mask = (const float*)d_in[1];
  const float* ln_w     = (const float*)d_in[2];
  const float* ln_b     = (const float*)d_in[3];
  const float* wq       = (const float*)d_in[4];
  const float* wk       = (const float*)d_in[5];
  const float* wvp      = (const float*)d_in[6];
  const float* wg       = (const float*)d_in[7];
  const float* bg       = (const float*)d_in[8];
  const float* wo       = (const float*)d_in[9];
  const float* bo       = (const float*)d_in[10];
  float* out            = (float*)d_out;

  // 320 KB dynamic LDS (per-WGP max on CDNA5); opt in every call
  // (deterministic; hipFuncSetAttribute is not a stream op, capture-safe).
  hipFuncSetAttribute(reinterpret_cast<const void*>(col_attn_fused),
                      hipFuncAttributeMaxDynamicSharedMemorySize,
                      LDS_TOTAL_BYTES);

  col_attn_fused<<<dim3(R_LEN), dim3(NTHREADS), LDS_TOTAL_BYTES, stream>>>(
      msa_act, msa_mask, ln_w, ln_b, wq, wk, wvp, wg, bg, wo, bo, out);
}
